// MoEsparseRoutingForClassification_80358838108606
// MI455X (gfx1250) — matrix operand order repaired and checked by reading, not compiled, not run on hardware
//
#include <hip/hip_runtime.h>
#include <math.h>

// CDNA5 WMMA vector types (wave32)
typedef __attribute__((ext_vector_type(16))) __bf16 v16bf;
typedef __attribute__((ext_vector_type(8)))  float  v8f;

#define B_  32
#define S_  128
#define H_  1024
#define E_  16
#define L_  8
#define KT  (E_ * H_)   // 16384 = fused contraction dim (e,h)

// Workspace layout:
//   [0          , 1 MB)          : Xg   bf16 [32][16384]  (gate-scaled activations)
//   [1 MB       , 1 MB + 128 KB) : Hpre f32  [32][1024]   (pre-tanh accumulator, init = gate-mixed bias)

// ---------------------------------------------------------------------------
// Kernel 1: Xg[b][e*1024+h] = bf16( gates[b][e] * X[b,0,h] )
// ---------------------------------------------------------------------------
__global__ void __launch_bounds__(256)
build_xg_kernel(const float* __restrict__ X, const float* __restrict__ gates,
                unsigned short* __restrict__ xg_raw) {
    __bf16* Xg = reinterpret_cast<__bf16*>(xg_raw);
    int idx = blockIdx.x * 256 + threadIdx.x;   // 0 .. 524287
    int b = idx >> 14;
    int e = (idx >> 10) & 15;
    int h = idx & 1023;
    float v = gates[b * E_ + e] * X[(size_t)b * S_ * H_ + h];  // X[:,0,:]
    Xg[idx] = (__bf16)v;
}

// ---------------------------------------------------------------------------
// Kernel 2: Hpre[b][o] = sum_e gates[b][e] * dense_b[e][o]
// ---------------------------------------------------------------------------
__global__ void __launch_bounds__(256)
init_hpre_kernel(const float* __restrict__ gates, const float* __restrict__ db,
                 float* __restrict__ Hpre) {
    int idx = blockIdx.x * 256 + threadIdx.x;   // 0 .. 32767
    int b = idx >> 10;
    int o = idx & 1023;
    float acc = 0.f;
#pragma unroll
    for (int e = 0; e < E_; ++e) acc += gates[b * E_ + e] * db[e * H_ + o];
    Hpre[idx] = acc;
}

// ---------------------------------------------------------------------------
// Kernel 3: split-K WMMA GEMM.  Hpre[b][o] += sum_h Xg[b][e*1024+h] * W[e][o][h]
// One wave = one 16x16 (MxN) tile for one expert's K-chunk of 1024 (32 WMMAs).
// 2048 wave-tiles total = 256 blocks x 8 waves.
// ---------------------------------------------------------------------------
__global__ void __launch_bounds__(256)
gemm_wmma_kernel(const unsigned short* __restrict__ xg_raw,
                 const float* __restrict__ W,
                 float* __restrict__ Hpre) {
    const __bf16* Xg = reinterpret_cast<const __bf16*>(xg_raw);

    const int lane  = threadIdx.x & 31;
    const int wave  = blockIdx.x * 8 + (threadIdx.x >> 5);  // 0 .. 2047
    const int e     = wave >> 7;        // expert = K split chunk
    const int otile = (wave >> 1) & 63; // N tile (o / 16)
    const int mtile = wave & 1;         // M tile (b / 16)

    const int nlo = lane & 15;          // A: row m ; B: col n
    const int hi  = lane >> 4;          // K-half selector

    // A fragment source: row (mtile*16+nlo) of Xg, K-chunk base e*1024, lane K-offset 8*hi.
    // Per 32-wide K step the lane needs bf16 [k+8*hi .. +7] and [k+8*hi+16 .. +23].
    const __bf16* pa = Xg + (size_t)(mtile * 16 + nlo) * KT + e * H_ + hi * 8;
    // B fragment source: dense_w row o = otile*16+nlo of expert e, f32 at k + 8*hi:
    // element i of the v16bf maps to k + 8*hi + i (i < 8) and k + 8*hi + 16 + (i - 8) (i >= 8).
    const float* pb = W + ((size_t)e * H_ + (otile * 16 + nlo)) * H_ + hi * 8;

    v8f c = {};
    for (int k = 0; k < H_; k += 32) {
        union { v16bf v; __bf16 h[16]; uint4 q[2]; } a;
        a.q[0] = *(const uint4*)(pa + k);        // 8 bf16 : K = k+off .. k+off+7
        a.q[1] = *(const uint4*)(pa + k + 16);   // 8 bf16 : K = k+off+16 .. +23

        union { v16bf v; __bf16 h[16]; } bm;
#pragma unroll
        for (int i = 0; i < 16; i += 4) {
            float4 f = *(const float4*)(pb + k + i + (i & 8));
            bm.h[i + 0] = (__bf16)f.x;
            bm.h[i + 1] = (__bf16)f.y;
            bm.h[i + 2] = (__bf16)f.z;
            bm.h[i + 3] = (__bf16)f.w;
        }
        // D = A x B + C  (16x16x32 bf16 -> f32), emits v_wmma_f32_16x16x32_bf16
        c = __builtin_amdgcn_wmma_f32_16x16x32_bf16(
                /*neg_a=*/false, a.v, /*neg_b=*/false, bm.v,
                /*c_mod=*/(short)0, c, /*reuse_a=*/false, /*reuse_b=*/false);
    }

    // C/D layout: VGPR r holds (m = r + 8*hi, n = nlo). Split-K reduce via f32 atomics.
    float* dst = Hpre + (size_t)(mtile * 16 + hi * 8) * H_ + otile * 16 + nlo;
#pragma unroll
    for (int r = 0; r < 8; ++r) atomicAdd(dst + (size_t)r * H_, c[r]);
}

// ---------------------------------------------------------------------------
// Kernel 4: out[b][l] = sum_h tanh(Hpre[b][h]) * (sum_e g[b][e]*ow[e][l][h])
//                       + sum_e g[b][e]*ob[e][l]
// One block per (b,l); 256-thread reduction over h. Tiny (4.5 MFLOP).
// ---------------------------------------------------------------------------
__global__ void __launch_bounds__(256)
out_proj_kernel(const float* __restrict__ Hpre, const float* __restrict__ gates,
                const float* __restrict__ ow, const float* __restrict__ ob,
                float* __restrict__ out) {
    __shared__ float red[256];
    const int b = blockIdx.x >> 3;
    const int l = blockIdx.x & 7;

    float g[E_];
#pragma unroll
    for (int e = 0; e < E_; ++e) g[e] = gates[b * E_ + e];

    float acc = 0.f;
    for (int h = threadIdx.x; h < H_; h += 256) {
        float hv = tanhf(Hpre[b * H_ + h]);
        float w = 0.f;
#pragma unroll
        for (int e = 0; e < E_; ++e) w += g[e] * ow[((size_t)e * L_ + l) * H_ + h];
        acc += hv * w;
    }
    red[threadIdx.x] = acc;
    __syncthreads();
#pragma unroll
    for (int s = 128; s > 0; s >>= 1) {
        if (threadIdx.x < s) red[threadIdx.x] += red[threadIdx.x + s];
        __syncthreads();
    }
    if (threadIdx.x == 0) {
        float bias = 0.f;
#pragma unroll
        for (int e = 0; e < E_; ++e) bias += g[e] * ob[e * L_ + l];
        out[blockIdx.x] = red[0] + bias;
    }
}

// ---------------------------------------------------------------------------
extern "C" void kernel_launch(void* const* d_in, const int* in_sizes, int n_in,
                              void* d_out, int out_size, void* d_ws, size_t ws_size,
                              hipStream_t stream) {
    (void)in_sizes; (void)n_in; (void)out_size; (void)ws_size;

    const float* X     = (const float*)d_in[0];  // (32,128,1024)
    const float* gates = (const float*)d_in[1];  // (32,16)
    const float* dw    = (const float*)d_in[2];  // (16,1024,1024)
    const float* db    = (const float*)d_in[3];  // (16,1024)
    const float* ow    = (const float*)d_in[4];  // (16,8,1024)
    const float* ob    = (const float*)d_in[5];  // (16,8)
    float*       out   = (float*)d_out;          // (32,8)

    unsigned short* Xg   = (unsigned short*)d_ws;                      // 1 MB bf16
    float*          Hpre = (float*)((char*)d_ws + (size_t)(1u << 20)); // 128 KB f32

    build_xg_kernel <<<2048, 256, 0, stream>>>(X, gates, Xg);      // 524288 elems
    init_hpre_kernel<<<128,  256, 0, stream>>>(gates, db, Hpre);   // 32768 elems
    gemm_wmma_kernel<<<256,  256, 0, stream>>>(Xg, dw, Hpre);      // 2048 wave-tiles
    out_proj_kernel <<<256,  256, 0, stream>>>(Hpre, gates, ow, ob, out);
}
